// BiLstmCrf_50800873177240
// MI455X (gfx1250) — matrix operand order repaired and checked by reading, not compiled, not run on hardware
//
#include <hip/hip_runtime.h>

// ---------------- problem constants ----------------
#define SEQ    512
#define BATCH  32
#define EMB    300
#define HID    150
#define TAGS   20
#define MROWS  (SEQ * BATCH)      // 16384 time-batch rows
#define NT     40                 // N tiles (600 padded to 640)
#define NCOL   (NT * 16)          // 640
#define KT_IN  10                 // K tiles for input GEMM (300 -> 320)
#define KT_H   5                  // K tiles for recurrent GEMM (150 -> 160)
#define ROWU   160                // u32 per packed f16 row (320 halves)

// LSTM LDS layout (dynamic shared, base offset 0: no static LDS in kernel)
#define HBUF_U32   (32 * 80)                       // packed f16 h state
#define CBUF_F32   (32 * 152)                      // f32 cell state
#define GIN_BYTES  (BATCH * NCOL * 4)              // 81920 B per gin buffer
#define GIN_OFF    (HBUF_U32 * 4 + CBUF_F32 * 4)   // 29696 B
#define LSTM_SMEM_BYTES (GIN_OFF + 2 * GIN_BYTES)  // 193536 B (<320KB/WGP)

typedef __attribute__((ext_vector_type(16))) _Float16 v16h;
typedef __attribute__((ext_vector_type(8)))  float    v8f;

union Frag16 { unsigned u[8]; v16h v; };
union HPair  { _Float16 h[2]; unsigned u; };
union HOne   { unsigned short s; _Float16 h; };

__device__ __forceinline__ float sigm(float x) { return 1.0f / (1.0f + __expf(-x)); }

// async global -> LDS b128 copy, tracked by ASYNCcnt. Same immediate offset is
// applied to both the LDS dest (VDST VGPR) and the global source (VADDR pair).
#define AS_CP(o) \
    asm volatile("global_load_async_to_lds_b128 %0, %1, off offset:" #o \
                 :: "v"(ldsdst), "v"(gaddr) : "memory");
#define ISSUE20 \
    AS_CP(0)     AS_CP(4096)  AS_CP(8192)  AS_CP(12288) AS_CP(16384) \
    AS_CP(20480) AS_CP(24576) AS_CP(28672) AS_CP(32768) AS_CP(36864) \
    AS_CP(40960) AS_CP(45056) AS_CP(49152) AS_CP(53248) AS_CP(57344) \
    AS_CP(61440) AS_CP(65536) AS_CP(69632) AS_CP(73728) AS_CP(77824)

// ---------------- utility: zero u32 buffer ----------------
__global__ void k_zero32(unsigned* p, int n) {
    int i = blockIdx.x * 256 + threadIdx.x;
    if (i < n) p[i] = 0u;
}

// ---------------- embedding gather -> packed f16 rows [MROWS][320] ----------------
__global__ void k_embed(const int* __restrict__ x, const float* __restrict__ emb,
                        unsigned* __restrict__ Xpack) {
    int row = blockIdx.x;          // s*32 + b
    int c2  = threadIdx.x;         // 0..159 (pair index)
    int id  = x[row];
    unsigned val = 0u;
    if (c2 < EMB / 2) {
        HPair p;
        p.h[0] = (_Float16)emb[(size_t)id * EMB + 2 * c2];
        p.h[1] = (_Float16)emb[(size_t)id * EMB + 2 * c2 + 1];
        val = p.u;
    }
    Xpack[(size_t)row * ROWU + c2] = val;
}

// ---------------- pack W[600][Kin] (row-major) into WMMA B-fragment layout ----------------
// frag[nt][kt][lane][8] u32 ; B[k][n] = W[n][k] ; lane: n = nt*16 + lane%16,
// half = lane/16, vgpr v holds K = kt*32 + (v<4 ? half*8 + 2v : 16 + half*8 + 2(v-4)) (+1 in hi 16 bits)
__global__ void k_pack_w(const float* __restrict__ W, unsigned* __restrict__ frag,
                         int Kin, int Ktiles) {
    int nt = blockIdx.x, kt = blockIdx.y;
    int lane = threadIdx.x;
    int half = lane >> 4, lm = lane & 15;
    int n = nt * 16 + lm;
    unsigned* dst = frag + ((size_t)(nt * Ktiles + kt)) * 256 + lane * 8;
#pragma unroll
    for (int v = 0; v < 8; ++v) {
        int k = kt * 32 + (v < 4 ? half * 8 + 2 * v : 16 + half * 8 + 2 * (v - 4));
        float f0 = (n < 4 * HID && k     < Kin) ? W[(size_t)n * Kin + k]     : 0.0f;
        float f1 = (n < 4 * HID && k + 1 < Kin) ? W[(size_t)n * Kin + k + 1] : 0.0f;
        HPair p; p.h[0] = (_Float16)f0; p.h[1] = (_Float16)f1;
        dst[v] = p.u;
    }
}

// ---------------- bias: bih + bhh, padded to 640 ----------------
__global__ void k_pack_bias(const float* __restrict__ bih, const float* __restrict__ bhh,
                            float* __restrict__ bias) {
    int n = threadIdx.x;
    bias[n] = (n < 4 * HID) ? (bih[n] + bhh[n]) : 0.0f;
}

// ---------------- big input GEMM: G[dir][16384][640] = Apack @ Wih^T + bias ----------------
// grid = (1024 m-tiles, 2 dirs), block = 256 (8 waves). Wave w owns nt = w + 8*i, i=0..4.
__global__ void k_gemm_in(const unsigned* __restrict__ Apack,   // [MROWS][ROWU]
                          const unsigned* __restrict__ Wfrag,   // [2][NT][KT_IN][32][8]
                          const float*    __restrict__ bias,    // [2][NCOL]
                          float*          __restrict__ G) {     // [2][MROWS][NCOL]
    int mt   = blockIdx.x;
    int dir  = blockIdx.y;
    int wave = threadIdx.x >> 5;
    int lane = threadIdx.x & 31;
    int half = lane >> 4, lm = lane & 15;

    const unsigned* wf = Wfrag + (size_t)dir * NT * KT_IN * 256;
    const float*    bs = bias + (size_t)dir * NCOL;
    float*          g  = G + (size_t)dir * MROWS * NCOL;

    v8f acc[5];
#pragma unroll
    for (int i = 0; i < 5; ++i)
#pragma unroll
        for (int e = 0; e < 8; ++e) acc[i][e] = 0.0f;

    int m = mt * 16 + lm;
#pragma unroll 1
    for (int kt = 0; kt < KT_IN; ++kt) {
        Frag16 a;
        int base = m * ROWU + kt * 16 + half * 4;
#pragma unroll
        for (int v = 0; v < 4; ++v) a.u[v] = Apack[base + v];
#pragma unroll
        for (int v = 0; v < 4; ++v) a.u[4 + v] = Apack[base + 8 + v];
#pragma unroll
        for (int i = 0; i < 5; ++i) {
            int nt = wave + 8 * i;
            Frag16 b;
            const unsigned* bp = wf + ((size_t)(nt * KT_IN + kt)) * 256 + lane * 8;
#pragma unroll
            for (int v = 0; v < 8; ++v) b.u[v] = bp[v];
            acc[i] = __builtin_amdgcn_wmma_f32_16x16x32_f16(
                false, a.v, false, b.v, (short)0, acc[i], false, false);
        }
    }
#pragma unroll
    for (int i = 0; i < 5; ++i) {
        int nt = wave + 8 * i;
        int n  = nt * 16 + lm;
        float bv = bs[n];
#pragma unroll
        for (int v = 0; v < 8; ++v) {
            int mr = mt * 16 + v + half * 8;
            g[(size_t)mr * NCOL + n] = acc[i][v] + bv;
        }
    }
}

// ---------------- persistent recurrent LSTM kernel ----------------
// grid = 2 (dir), block = 256 (8 waves). Dynamic LDS:
//   hbuf: u32 [32][80]       packed f16 h (K padded to 160)
//   cbuf: f32 [32][152]      cell state
//   ginb: f32 2x[32][640]    ping-pong gin staging (async-prefetched),
//                            gate preactivations written back in place.
__global__ void k_lstm(const float*    __restrict__ G,      // [2][MROWS][NCOL]
                       const unsigned* __restrict__ WhhF,   // [2][NT][KT_H][32][8]
                       unsigned*       __restrict__ Hpack) {// [MROWS][ROWU]
    extern __shared__ unsigned char smem[];
    unsigned* hbuf = (unsigned*)smem;                    // 32*80 u32
    float*    cbuf = (float*)(smem + HBUF_U32 * 4);      // 32*152 f32
    float*    ginb = (float*)(smem + GIN_OFF);           // 2 x 32*640 f32

    int dir  = blockIdx.x;
    int tid  = threadIdx.x;
    int wave = tid >> 5;
    int lane = tid & 31;
    int half = lane >> 4, lm = lane & 15;

    for (int i = tid; i < HBUF_U32; i += 256) hbuf[i] = 0u;
    for (int i = tid; i < CBUF_F32; i += 256) cbuf[i] = 0.0f;
    __syncthreads();

    const float* Gd = G + (size_t)dir * MROWS * NCOL;

    // prologue: async-fill buffer 0 with gin(step 0)
    {
        int s0 = dir ? (SEQ - 1) : 0;
        unsigned ldsdst = (unsigned)(GIN_OFF + tid * 16);
        unsigned long long gaddr =
            (unsigned long long)(Gd + (size_t)s0 * BATCH * NCOL) + (unsigned)(tid * 16);
        ISSUE20;
    }

    for (int step = 0; step < SEQ; ++step) {
        int s = dir ? (SEQ - 1 - step) : step;
        int p = step & 1;

        // wait for buf[p]'s async fill (issued previous iteration / prologue)
        asm volatile("s_wait_asynccnt 0x0" ::: "memory");
        __syncthreads();

        // kick off prefetch of next step's gin into buf[1-p]
        if (step + 1 < SEQ) {
            int sn = dir ? (SEQ - 2 - step) : (step + 1);
            unsigned ldsdst = (unsigned)(GIN_OFF + (1 - p) * GIN_BYTES + tid * 16);
            unsigned long long gaddr =
                (unsigned long long)(Gd + (size_t)sn * BATCH * NCOL) + (unsigned)(tid * 16);
            ISSUE20;
        }

        float* gc = ginb + p * (GIN_BYTES / 4);

        // Phase A: gates = h @ Whh^T + gin  (80 tiles across 8 waves, in-place)
#pragma unroll 1
        for (int i = 0; i < 10; ++i) {
            int t2 = wave + 8 * i;        // 0..79
            int mt = t2 / NT, nt = t2 % NT;
            v8f acc;
#pragma unroll
            for (int e = 0; e < 8; ++e) acc[e] = 0.0f;
            int m = mt * 16 + lm;
#pragma unroll
            for (int kt = 0; kt < KT_H; ++kt) {
                Frag16 a, b;
                int base = m * 80 + kt * 16 + half * 4;
#pragma unroll
                for (int v = 0; v < 4; ++v) a.u[v] = hbuf[base + v];
#pragma unroll
                for (int v = 0; v < 4; ++v) a.u[4 + v] = hbuf[base + 8 + v];
                const unsigned* bp =
                    WhhF + (((size_t)dir * NT + nt) * KT_H + kt) * 256 + lane * 8;
#pragma unroll
                for (int v = 0; v < 8; ++v) b.u[v] = bp[v];
                acc = __builtin_amdgcn_wmma_f32_16x16x32_f16(
                    false, a.v, false, b.v, (short)0, acc, false, false);
            }
            int n = nt * 16 + lm;
#pragma unroll
            for (int v = 0; v < 8; ++v) {
                int mr = mt * 16 + v + half * 8;            // batch row 0..31
                gc[mr * NCOL + n] += acc[v];                // fuse gin + h@Whh^T in place
            }
        }
        __syncthreads();

        // Phase B: nonlinearities + state update (2400 f16 pairs)
        for (int e = tid; e < BATCH * (HID / 2); e += 256) {
            int b = e / (HID / 2), jp = e % (HID / 2), j = 2 * jp;
            const float* gr = gc + b * NCOL;
            float i0 = gr[j],           i1 = gr[j + 1];
            float f0 = gr[HID + j],     f1 = gr[HID + j + 1];
            float g0 = gr[2 * HID + j], g1 = gr[2 * HID + j + 1];
            float o0 = gr[3 * HID + j], o1 = gr[3 * HID + j + 1];
            float c0 = sigm(f0) * cbuf[b * 152 + j]     + sigm(i0) * tanhf(g0);
            float c1 = sigm(f1) * cbuf[b * 152 + j + 1] + sigm(i1) * tanhf(g1);
            cbuf[b * 152 + j]     = c0;
            cbuf[b * 152 + j + 1] = c1;
            HPair hp;
            hp.h[0] = (_Float16)(sigm(o0) * tanhf(c0));
            hp.h[1] = (_Float16)(sigm(o1) * tanhf(c1));
            hbuf[b * 80 + jp] = hp.u;
            Hpack[((size_t)s * BATCH + b) * ROWU + dir * (HID / 2) + jp] = hp.u;
        }
        __syncthreads();
    }
}

// ---------------- emissions: em[row][t] = Hcat1[row] . Wout[t] + bout[t] ----------------
__global__ void k_emis(const unsigned short* __restrict__ H,   // [MROWS][320] f16
                       const float* __restrict__ Wout,          // [TAGS][300]
                       const float* __restrict__ bout,
                       float* __restrict__ em) {                // [MROWS][TAGS]
    int row = blockIdx.x;
    int t   = threadIdx.x;
    if (t < TAGS) {
        float acc = bout[t];
        const unsigned short* h = H + (size_t)row * (2 * ROWU);
        for (int c = 0; c < 2 * HID; ++c) {
            HOne cv; cv.s = h[c];
            acc += (float)cv.h * Wout[t * (2 * HID) + c];
        }
        em[(size_t)row * TAGS + t] = acc;
    }
}

// ---------------- Viterbi forward + backtrack (single block, 640 threads) ----------------
__global__ void k_viterbi(const float* __restrict__ em,    // [SEQ][BATCH][TAGS]
                          const float* __restrict__ start,
                          const float* __restrict__ endt,
                          const float* __restrict__ trans, // [TAGS][TAGS]
                          unsigned char* __restrict__ hist,// [SEQ-1][BATCH][TAGS]
                          float* __restrict__ out) {       // tags[32][512] then best[32]
    __shared__ float sc[BATCH * TAGS];
    __shared__ float tr[TAGS * TAGS];
    int tid = threadIdx.x;
    int b = tid / TAGS, t = tid % TAGS;
    if (tid < TAGS * TAGS) tr[tid] = trans[tid];
    sc[tid] = start[t] + em[tid];
    __syncthreads();
    for (int s = 1; s < SEQ; ++s) {
        float e = em[(size_t)s * BATCH * TAGS + tid];
        float best = -3.4e38f; int arg = 0;
#pragma unroll
        for (int tp = 0; tp < TAGS; ++tp) {
            float v = sc[b * TAGS + tp] + tr[tp * TAGS + t];
            if (v > best) { best = v; arg = tp; }
        }
        hist[(size_t)(s - 1) * BATCH * TAGS + tid] = (unsigned char)arg;
        __syncthreads();
        sc[tid] = best + e;
        __syncthreads();
    }
    sc[tid] += endt[t];
    __syncthreads();
    if (tid < BATCH) {
        float bv = sc[tid * TAGS]; int bt = 0;
        for (int t2 = 1; t2 < TAGS; ++t2) {
            float v = sc[tid * TAGS + t2];
            if (v > bv) { bv = v; bt = t2; }
        }
        out[(size_t)SEQ * BATCH + tid] = bv;
        int tag = bt;
        out[(size_t)tid * SEQ + (SEQ - 1)] = (float)tag;
        for (int s = SEQ - 2; s >= 0; --s) {
            tag = hist[(size_t)s * BATCH * TAGS + tid * TAGS + tag];
            out[(size_t)tid * SEQ + s] = (float)tag;
        }
    }
}

// ---------------- host orchestration ----------------
extern "C" void kernel_launch(void* const* d_in, const int* in_sizes, int n_in,
                              void* d_out, int out_size, void* d_ws, size_t ws_size,
                              hipStream_t stream) {
    (void)in_sizes; (void)n_in; (void)out_size; (void)ws_size;

    const int*   x    = (const int*)d_in[0];
    const float* emb  = (const float*)d_in[2];
    const float* Wih[2][2], *Whh[2][2], *bih[2][2], *bhh[2][2];
    for (int l = 0; l < 2; ++l)
        for (int d = 0; d < 2; ++d) {
            int base = 3 + (l * 2 + d) * 4;
            Wih[l][d] = (const float*)d_in[base + 0];
            Whh[l][d] = (const float*)d_in[base + 1];
            bih[l][d] = (const float*)d_in[base + 2];
            bhh[l][d] = (const float*)d_in[base + 3];
        }
    const float* Wout   = (const float*)d_in[19];
    const float* bout   = (const float*)d_in[20];
    const float* startt = (const float*)d_in[21];
    const float* endt   = (const float*)d_in[22];
    const float* trans  = (const float*)d_in[23];
    float* out = (float*)d_out;

    // workspace carve-out
    size_t off = 0;
    auto alloc = [&](size_t bytes) -> char* {
        char* p = (char*)d_ws + off;
        off += (bytes + 255) & ~(size_t)255;
        return p;
    };
    unsigned* Xpack = (unsigned*)alloc((size_t)MROWS * ROWU * 4);
    unsigned* Hc0   = (unsigned*)alloc((size_t)MROWS * ROWU * 4);
    unsigned* Hc1   = (unsigned*)alloc((size_t)MROWS * ROWU * 4);
    float*    G     = (float*)   alloc((size_t)2 * MROWS * NCOL * 4);
    unsigned* WihF  = (unsigned*)alloc((size_t)4 * NT * KT_IN * 256 * 4);
    unsigned* WhhF  = (unsigned*)alloc((size_t)4 * NT * KT_H * 256 * 4);
    float*    biasv = (float*)   alloc((size_t)4 * NCOL * 4);
    float*    em    = (float*)   alloc((size_t)MROWS * TAGS * 4);
    unsigned char* hist = (unsigned char*)alloc((size_t)(SEQ - 1) * BATCH * TAGS);

    const size_t WIHF_D = (size_t)NT * KT_IN * 256;   // u32 per (layer,dir)
    const size_t WHHF_D = (size_t)NT * KT_H * 256;

    // zero Hcat buffers (pad columns must stay zero for next layer's GEMM)
    {
        int n = MROWS * ROWU;
        k_zero32<<<dim3((n + 255) / 256), dim3(256), 0, stream>>>(Hc0, n);
        k_zero32<<<dim3((n + 255) / 256), dim3(256), 0, stream>>>(Hc1, n);
    }

    // embedding gather + weight packing
    k_embed<<<dim3(MROWS), dim3(ROWU), 0, stream>>>(x, emb, Xpack);
    for (int l = 0; l < 2; ++l)
        for (int d = 0; d < 2; ++d) {
            int idx = l * 2 + d;
            k_pack_w<<<dim3(NT, KT_IN), dim3(32), 0, stream>>>(
                Wih[l][d], WihF + (size_t)idx * WIHF_D, (l == 0) ? EMB : 2 * HID, KT_IN);
            k_pack_w<<<dim3(NT, KT_H), dim3(32), 0, stream>>>(
                Whh[l][d], WhhF + (size_t)idx * WHHF_D, HID, KT_H);
            k_pack_bias<<<dim3(1), dim3(NCOL), 0, stream>>>(
                bih[l][d], bhh[l][d], biasv + (size_t)idx * NCOL);
        }

    // layer 0
    k_gemm_in<<<dim3(MROWS / 16, 2), dim3(256), 0, stream>>>(Xpack, WihF, biasv, G);
    k_lstm<<<dim3(2), dim3(256), LSTM_SMEM_BYTES, stream>>>(G, WhhF, Hc0);

    // layer 1
    k_gemm_in<<<dim3(MROWS / 16, 2), dim3(256), 0, stream>>>(
        Hc0, WihF + 2 * WIHF_D, biasv + 2 * NCOL, G);
    k_lstm<<<dim3(2), dim3(256), LSTM_SMEM_BYTES, stream>>>(G, WhhF + 2 * WHHF_D, Hc1);

    // emissions + Viterbi
    k_emis<<<dim3(MROWS), dim3(32), 0, stream>>>((const unsigned short*)Hc1, Wout, bout, em);
    k_viterbi<<<dim3(1), dim3(BATCH * TAGS), 0, stream>>>(em, startt, endt, trans, hist, out);
}
